// MultitaskDebateGNN_52853867544811
// MI455X (gfx1250) — compile-verified, matrix operand order checked by reading, not run on hardware
//
#include <hip/hip_runtime.h>
#include <math.h>

typedef __attribute__((ext_vector_type(16))) _Float16 v16h;
typedef __attribute__((ext_vector_type(8)))  float    v8f;

__device__ __forceinline__ float gelu_exact(float v) {
    return 0.5f * v * (1.0f + erff(v * 0.7071067811865476f));
}

__global__ void zero_kernel(float* __restrict__ p, int n) {
    int i = blockIdx.x * blockDim.x + threadIdx.x;
    int stride = gridDim.x * blockDim.x;
    for (; i < n; i += stride) p[i] = 0.0f;
}

__global__ void half_prep_kernel(const float* __restrict__ w, _Float16* __restrict__ h, int n) {
    int i = blockIdx.x * blockDim.x + threadIdx.x;
    if (i < n) h[i] = (_Float16)w[i];
}

__global__ void count_kernel(const int* __restrict__ dst, float* __restrict__ cnt, int E) {
    int e = blockIdx.x * blockDim.x + threadIdx.x;
    if (e < E) atomicAdd(&cnt[dst[e]], 1.0f);
}

// One block = 16 edges. 8 wave32 per block.
// Wave w handles n-tiles t = w + 8j (j=0..7): fixed o-half = w&1, i = t>>1.
__global__ __launch_bounds__(256) void edge_kernel(
    const float* __restrict__ x,          // node features (layer input), [N][32]
    const int*   __restrict__ src_idx,    // [E]
    const int*   __restrict__ dst_idx,    // [E]
    const float* __restrict__ eattr,      // [E][4] = conf(1) + stance(3)
    const float* __restrict__ vh_w,       // [16][3]
    const float* __restrict__ vh_b,       // [16]
    const float* __restrict__ e1_w,       // [128][17]
    const float* __restrict__ e1_b,       // [128]
    const float* __restrict__ ln_g,       // [128]
    const float* __restrict__ ln_b,       // [128]
    const _Float16* __restrict__ e2w_h,   // [1024][128] f16 copy of e2_w
    const float* __restrict__ e2_b,       // [1024]
    float*       __restrict__ agg,        // [N][32] accumulated messages
    int E)
{
    __shared__ float    seL[16][16];      // gelu(stance @ vh_w^T + vh_b)
    __shared__ float    h32[16][128];     // e1 output (pre-LN)
    __shared__ _Float16 h16[16][128];     // LN+gelu, f16, WMMA A source
    __shared__ float    xsl[16][32];      // x[src] tile
    __shared__ float    muL[16], rsL[16];
    __shared__ float    ssAll[16];        // per-edge sum of squares of w
    __shared__ float    rnL[16];          // 1 / max(||w||, 1e-6)
    __shared__ float    msgbuf[2][16][16];// [o-half][edge][o%16]

    const int tid = threadIdx.x;
    const int eb  = blockIdx.x * 16;

    // ---------------- Phase 1a: se, x[src] tile, zero LDS accumulators ----
    {
        int el = tid & 15;
        int j  = tid >> 4;
        int e  = eb + el; if (e >= E) e = eb;
        float c0 = eattr[e * 4 + 1];
        float c1 = eattr[e * 4 + 2];
        float c2 = eattr[e * 4 + 3];
        float v = vh_b[j] + c0 * vh_w[j * 3 + 0] + c1 * vh_w[j * 3 + 1] + c2 * vh_w[j * 3 + 2];
        seL[el][j] = gelu_exact(v);
    }
    #pragma unroll
    for (int q = 0; q < 2; ++q) {
        int idx = tid + 256 * q;           // 0..511
        int el = idx >> 5, i = idx & 31;
        int e = eb + el; if (e >= E) e = eb;
        xsl[el][i] = x[(size_t)src_idx[e] * 32 + i];
        ((float*)msgbuf)[idx] = 0.0f;
    }
    if (tid < 16) ssAll[tid] = 0.0f;
    __syncthreads();

    // ---------------- Phase 1b: h = [conf|se] @ e1_w^T + e1_b -------------
    {
        int el = tid & 15;
        int g  = tid >> 4;
        int e  = eb + el; if (e >= E) e = eb;
        float conf = eattr[e * 4 + 0];
        for (int q = 0; q < 8; ++q) {
            int k = g * 8 + q;
            float s = e1_b[k] + conf * e1_w[k * 17 + 0];
            #pragma unroll
            for (int j = 0; j < 16; ++j) s += seL[el][j] * e1_w[k * 17 + 1 + j];
            h32[el][k] = s;
        }
    }
    __syncthreads();

    // ---------------- Phase 1c: LayerNorm stats ---------------------------
    if (tid < 16) {
        float mu = 0.0f;
        for (int k = 0; k < 128; ++k) mu += h32[tid][k];
        mu *= (1.0f / 128.0f);
        float var = 0.0f;
        for (int k = 0; k < 128; ++k) { float d = h32[tid][k] - mu; var += d * d; }
        var *= (1.0f / 128.0f);
        muL[tid] = mu;
        rsL[tid] = rsqrtf(var + 1e-5f);
    }
    __syncthreads();

    // ---------------- Phase 1d: normalize + gelu + f16 --------------------
    {
        int el = tid & 15;
        int g  = tid >> 4;
        float mu = muL[el], rs = rsL[el];
        for (int q = 0; q < 8; ++q) {
            int k = g * 8 + q;
            float hn = (h32[el][k] - mu) * rs * ln_g[k] + ln_b[k];
            h16[el][k] = (_Float16)gelu_exact(hn);
        }
    }
    __syncthreads();

    // ---------------- Phase 2: WMMA GEMM (16x128)x(128x1024) --------------
    const int wv   = tid >> 5;       // wave id 0..7
    const int lane = tid & 31;
    const int half = lane >> 4;      // 0: lanes 0-15, 1: lanes 16-31
    const int lm   = lane & 15;      // A: row M; B/C: column N(mod 16)

    float wreg[8][8];                // biased w values: [j][C-vgpr]
    float ss[8];
    #pragma unroll
    for (int r = 0; r < 8; ++r) ss[r] = 0.0f;

    #pragma unroll
    for (int j = 0; j < 8; ++j) {
        const int t = wv + 8 * j;    // global n-tile 0..63
        const int nBase = 16 * t;
        v8f acc = {};
        #pragma unroll
        for (int ks = 0; ks < 4; ++ks) {
            const int kb = 32 * ks;
            union { v16h v; unsigned int u[8]; } a, b;
            #pragma unroll
            for (int v = 0; v < 8; ++v) {
                // 16-bit A/B VGPR layout: VGPR v holds K pair (k0, k0+1)
                int k0 = kb + 2 * v + 8 * half + ((v >= 4) ? 8 : 0);
                a.u[v] = *(const unsigned int*)&h16[lm][k0];
                b.u[v] = *(const unsigned int*)&e2w_h[(size_t)(nBase + lm) * 128 + k0];
            }
            acc = __builtin_amdgcn_wmma_f32_16x16x32_f16(
                false, a.v, false, b.v, (short)0, acc, false, false);
        }
        float bb = e2_b[nBase + lm];
        #pragma unroll
        for (int r = 0; r < 8; ++r) {
            float wval = acc[r] + bb;     // row M = r + 8*half, col = nBase + lm
            wreg[j][r] = wval;
            ss[r] += wval * wval;
        }
    }

    // reduce sum-of-squares across the 16 lanes of each half (row-wise)
    #pragma unroll
    for (int r = 0; r < 8; ++r) {
        float s = ss[r];
        s += __shfl_xor(s, 1, 32);
        s += __shfl_xor(s, 2, 32);
        s += __shfl_xor(s, 4, 32);
        s += __shfl_xor(s, 8, 32);
        ss[r] = s;
    }
    if (lm == 0) {
        #pragma unroll
        for (int r = 0; r < 8; ++r) atomicAdd(&ssAll[r + 8 * half], ss[r]);
    }

    // per-edge message partials: msg[e][o] = sum_i x[src[e]][i] * w[e][i*32+o]
    float msg[8];
    #pragma unroll
    for (int r = 0; r < 8; ++r) msg[r] = 0.0f;
    #pragma unroll
    for (int j = 0; j < 8; ++j) {
        const int t = wv + 8 * j;
        const int i = t >> 1;             // input feature index of this tile
        #pragma unroll
        for (int r = 0; r < 8; ++r)
            msg[r] += xsl[r + 8 * half][i] * wreg[j][r];
    }
    __syncthreads();
    if (tid < 16) rnL[tid] = 1.0f / fmaxf(sqrtf(ssAll[tid]), 1e-6f);
    __syncthreads();
    {
        const int oh = wv & 1;            // output half this wave covers
        #pragma unroll
        for (int r = 0; r < 8; ++r)
            atomicAdd(&msgbuf[oh][r + 8 * half][lm], msg[r] * rnL[r + 8 * half]);
    }
    __syncthreads();

    // ---------------- Phase 3: scatter to agg[dst] ------------------------
    #pragma unroll
    for (int q = 0; q < 2; ++q) {
        int idx = tid + 256 * q;          // 0..511
        int el = idx >> 5, o = idx & 31;
        int e = eb + el;
        if (e < E)
            atomicAdd(&agg[(size_t)dst_idx[e] * 32 + o], msgbuf[o >> 4][el][o & 15]);
    }
}

// out[n][o] = gelu( agg[n][o]/max(cnt[n],1) + x[n]·root_w[o] + b[o] )
__global__ void node_kernel(const float* __restrict__ x, const float* __restrict__ agg,
                            const float* __restrict__ cnt, const float* __restrict__ root_w,
                            const float* __restrict__ bias, float* __restrict__ out, int N)
{
    int idx = blockIdx.x * blockDim.x + threadIdx.x;
    if (idx >= N * 32) return;
    int n = idx >> 5, o = idx & 31;
    const float* xr = x + (size_t)n * 32;
    const float* wr = root_w + o * 32;
    float s = bias[o];
    #pragma unroll
    for (int i = 0; i < 32; ++i) s += xr[i] * wr[i];
    float a = agg[idx] / fmaxf(cnt[n], 1.0f);
    out[idx] = gelu_exact(a + s);
}

extern "C" void kernel_launch(void* const* d_in, const int* in_sizes, int n_in,
                              void* d_out, int out_size, void* d_ws, size_t ws_size,
                              hipStream_t stream) {
    const float* x     = (const float*)d_in[0];
    const int*   ei    = (const int*)d_in[1];
    const float* eattr = (const float*)d_in[2];
    // layer 0 params: indices 3..12 ; layer 1 params: 13..22
    const float* vh_w0 = (const float*)d_in[3];
    const float* vh_b0 = (const float*)d_in[4];
    const float* e1_w0 = (const float*)d_in[5];
    const float* e1_b0 = (const float*)d_in[6];
    const float* ln_g0 = (const float*)d_in[7];
    const float* ln_b0 = (const float*)d_in[8];
    const float* e2_w0 = (const float*)d_in[9];
    const float* e2_b0 = (const float*)d_in[10];
    const float* rw0   = (const float*)d_in[11];
    const float* b0    = (const float*)d_in[12];
    const float* vh_w1 = (const float*)d_in[13];
    const float* vh_b1 = (const float*)d_in[14];
    const float* e1_w1 = (const float*)d_in[15];
    const float* e1_b1 = (const float*)d_in[16];
    const float* ln_g1 = (const float*)d_in[17];
    const float* ln_b1 = (const float*)d_in[18];
    const float* e2_w1 = (const float*)d_in[19];
    const float* e2_b1 = (const float*)d_in[20];
    const float* rw1   = (const float*)d_in[21];
    const float* b1    = (const float*)d_in[22];

    const int N = in_sizes[0] / 32;
    const int E = in_sizes[1] / 2;
    const int* src = ei;
    const int* dst = ei + E;

    // workspace layout (contiguous, float-aligned):
    float*    agg    = (float*)d_ws;                     // N*32
    float*    cnt    = agg + (size_t)N * 32;             // N
    float*    hnodes = cnt + N;                          // N*32
    _Float16* e2h0   = (_Float16*)(hnodes + (size_t)N * 32); // 1024*128
    _Float16* e2h1   = e2h0 + 1024 * 128;                    // 1024*128

    const int W2 = 1024 * 128;
    half_prep_kernel<<<(W2 + 255) / 256, 256, 0, stream>>>(e2_w0, e2h0, W2);
    half_prep_kernel<<<(W2 + 255) / 256, 256, 0, stream>>>(e2_w1, e2h1, W2);

    // zero agg + cnt (contiguous: N*33 floats)
    zero_kernel<<<1024, 256, 0, stream>>>(agg, N * 33);
    count_kernel<<<(E + 255) / 256, 256, 0, stream>>>(dst, cnt, E);

    const int egrid = (E + 15) / 16;
    const int ngrid = (N * 32 + 255) / 256;

    // ---- layer 0 ----
    edge_kernel<<<egrid, 256, 0, stream>>>(x, src, dst, eattr,
        vh_w0, vh_b0, e1_w0, e1_b0, ln_g0, ln_b0, e2h0, e2_b0, agg, E);
    node_kernel<<<ngrid, 256, 0, stream>>>(x, agg, cnt, rw0, b0, hnodes, N);

    // ---- layer 1 ----
    zero_kernel<<<1024, 256, 0, stream>>>(agg, N * 32);
    edge_kernel<<<egrid, 256, 0, stream>>>(hnodes, src, dst, eattr,
        vh_w1, vh_b1, e1_w1, e1_b1, ln_g1, ln_b1, e2h1, e2_b1, agg, E);
    node_kernel<<<ngrid, 256, 0, stream>>>(hnodes, agg, cnt, rw1, b1, (float*)d_out, N);
}
